// Category_TopK_Pooling_15049565405715
// MI455X (gfx1250) — matrix operand order repaired and checked by reading, not compiled, not run on hardware
//
#include <hip/hip_runtime.h>
#include <hip/hip_bf16.h>
#include <math.h>

typedef __attribute__((ext_vector_type(16))) _Float16 v16h;
typedef __attribute__((ext_vector_type(8)))  float    v8f;
typedef __attribute__((ext_vector_type(4)))  float    v4f;

#define MM    90
#define KKEEP 45
#define TENL  23
#define TENR  22

// ---------------- kernel 0: ||w|| ----------------
__global__ __launch_bounds__(128) void k_wnorm(const float* __restrict__ w, int D,
                                               float* __restrict__ wnorm) {
    __shared__ float red[128];
    float s = 0.f;
    for (int i = threadIdx.x; i < D; i += 128) { float v = w[i]; s += v * v; }
    red[threadIdx.x] = s;
    __syncthreads();
    for (int off = 64; off > 0; off >>= 1) {
        if (threadIdx.x < off) red[threadIdx.x] += red[threadIdx.x + off];
        __syncthreads();
    }
    if (threadIdx.x == 0) *wnorm = sqrtf(red[0]);
}

// ---------------- kernel 1: score = tanh(x.w / ||w||) via WMMA ----------------
// One wave per 16-row tile. A = 16x32 f16 chunk of x, B = w chunk broadcast to
// all 16 columns (lane l holds column l%16 with the same k order as its A row, so a
// broadcast B is just b[i] = w[k of a[i]]). x loads stay regular-
// temporal on purpose: the gather pass re-reads ~half of x and L2 is 192 MB.
__global__ __launch_bounds__(256) void k_scores(const float* __restrict__ x,
                                                const float* __restrict__ w,
                                                const float* __restrict__ wnorm,
                                                float* __restrict__ scores,
                                                long N, int D) {
    __shared__ _Float16 wh[512];
    for (int i = threadIdx.x; i < D; i += 256) wh[i] = (_Float16)w[i];
    __syncthreads();

    const float inv_nw = 1.0f / *wnorm;
    const int wave = threadIdx.x >> 5;
    const int lane = threadIdx.x & 31;
    const long tile = (long)blockIdx.x * 8 + wave;
    const long r0   = tile * 16;
    if (r0 >= N) return;                       // wave-uniform: EXEC stays all-ones

    const int mrow = lane & 15;                // A-fragment row for this lane
    const int hi   = lane >> 4;                // 0: K 0-7/16-23, 1: K 8-15/24-31
    const int koff = hi * 8;
    const float* xrow = x + (r0 + mrow) * (long)D;

    v8f c = {};
    for (int k0 = 0; k0 < D; k0 += 32) {
        const v4f* p0 = (const v4f*)(xrow + k0 + koff);
        const v4f* p1 = (const v4f*)(xrow + k0 + koff + 16);
        v4f f0 = p0[0];
        v4f f1 = p0[1];
        v4f f2 = p1[0];
        v4f f3 = p1[1];
        v16h a;
        a[0]  = (_Float16)f0.x; a[1]  = (_Float16)f0.y;
        a[2]  = (_Float16)f0.z; a[3]  = (_Float16)f0.w;
        a[4]  = (_Float16)f1.x; a[5]  = (_Float16)f1.y;
        a[6]  = (_Float16)f1.z; a[7]  = (_Float16)f1.w;
        a[8]  = (_Float16)f2.x; a[9]  = (_Float16)f2.y;
        a[10] = (_Float16)f2.z; a[11] = (_Float16)f2.w;
        a[12] = (_Float16)f3.x; a[13] = (_Float16)f3.y;
        a[14] = (_Float16)f3.z; a[15] = (_Float16)f3.w;

        v16h b;
#pragma unroll
        for (int i = 0; i < 16; ++i) b[i] = wh[k0 + koff + (i < 8 ? i : i + 8)];

        c = __builtin_amdgcn_wmma_f32_16x16x32_f16(
                /*neg_a=*/false, a, /*neg_b=*/false, b,
                /*c_mod=*/(short)0, c, /*reuse_a=*/false, /*reuse_b=*/false);
    }

    // All 16 output columns are equal; lane 0 holds rows r0..r0+7 in c[0..7],
    // lane 16 holds rows r0+8..r0+15.
    if (mrow == 0) {
        long base = r0 + (long)hi * 8;
#pragma unroll
        for (int r = 0; r < 8; ++r)
            scores[base + r] = tanhf(c[r] * inv_nw);
    }
}

// ---------------- kernel 2: per-graph sort + category top-K ----------------
__global__ __launch_bounds__(128) void k_pool(const float* __restrict__ scores,
                                              int* __restrict__ node_map,
                                              int* __restrict__ perm_int,
                                              float* __restrict__ out_batch,
                                              float* __restrict__ out_perm,
                                              float* __restrict__ out_perm_lab) {
    const int b = blockIdx.x;
    __shared__ float sc[MM];
    __shared__ int   order[MM];
    __shared__ int   par[MM];
    const int t = threadIdx.x;

    if (t < MM) {
        sc[t] = scores[(long)b * MM + t];
        node_map[(long)b * MM + t] = -1;       // block-local slice of node_map
    }
    __syncthreads();

    if (t < MM) {                              // stable rank for argsort(-s)
        float si = sc[t];
        int cnt = 0;
        for (int j = 0; j < MM; ++j) {
            float sj = sc[j];
            cnt += (sj > si) || (sj == si && j < t);
        }
        order[cnt] = t;
    }
    __syncthreads();
    if (t < MM) par[t] = order[t] & 1;
    __syncthreads();

    if (t < MM) {
        int ecnt = 0;
        for (int q = 0; q < t; ++q) ecnt += (par[q] == 0);
        int ocnt = t - ecnt;
        int nid  = order[t];
        int g    = b * MM + nid;
        out_perm_lab[(long)b * MM + t] = (float)g;

        int slot = -1;
        if ((nid & 1) == 0) { if (ecnt < TENL) slot = ecnt; }
        else                { if (ocnt < TENR) slot = TENL + ocnt; }
        if (slot >= 0) {
            int orow = b * KKEEP + slot;
            perm_int[orow]  = g;
            node_map[g]     = orow;
            out_perm[orow]  = (float)g;
            out_batch[orow] = (float)b;
        }
    }
}

// ---------------- kernel 3: x_out = x[perm] * score[perm] ----------------
// x loads: regular temporal (want the L2-resident copy from k_scores).
// x_out stores: non-temporal — 189 MB write-only stream, never re-read.
__global__ __launch_bounds__(256) void k_gather(const float* __restrict__ x,
                                                const float* __restrict__ scores,
                                                const int* __restrict__ perm_int,
                                                float* __restrict__ x_out, int D) {
    long gid = (long)blockIdx.x * blockDim.x + threadIdx.x;
    int  vpr = D >> 2;                          // float4 per row
    long r   = gid / vpr;
    int  cc  = (int)(gid - r * vpr);
    int  g   = perm_int[r];
    float s  = scores[g];
    const v4f* src = (const v4f*)(x + (long)g * D);
    v4f v = src[cc];
    v4f o = v * s;
    __builtin_nontemporal_store(o, ((v4f*)(x_out + r * (long)D)) + cc);
}

// ---------------- kernel 4: edge relabel + mask ----------------
// edge_index: NT load (32 MB streamed once). node_map: RT (737 KB, ~8M random
// hits — keep L2-resident). Outputs: NT stores (48 MB write-only stream).
__global__ __launch_bounds__(256) void k_edges(const int* __restrict__ ei,
                                               const int* __restrict__ node_map,
                                               long E,
                                               float* __restrict__ out_edge,
                                               float* __restrict__ out_mask) {
    long e = (long)blockIdx.x * blockDim.x + threadIdx.x;
    if (e >= E) return;
    int sn = __builtin_nontemporal_load(ei + e);
    int dn = __builtin_nontemporal_load(ei + E + e);
    int r = node_map[sn];
    int c = node_map[dn];
    bool m = (r >= 0) && (c >= 0);
    __builtin_nontemporal_store(m ? (float)r : -1.0f, out_edge + e);
    __builtin_nontemporal_store(m ? (float)c : -1.0f, out_edge + E + e);
    __builtin_nontemporal_store(m ? 1.0f : 0.0f,      out_mask + e);
}

extern "C" void kernel_launch(void* const* d_in, const int* in_sizes, int n_in,
                              void* d_out, int out_size, void* d_ws, size_t ws_size,
                              hipStream_t stream) {
    const float* x  = (const float*)d_in[0];
    const float* w  = (const float*)d_in[1];
    const int*   ei = (const int*)d_in[2];
    (void)n_in; (void)out_size; (void)ws_size;   // batch input unused (batch_out = perm / 90)

    const int  D = in_sizes[1];
    const long N = (long)in_sizes[0] / D;
    const long E = (long)in_sizes[2] / 2;
    const int  B = (int)(N / MM);

    // d_out: [x_out B*45*D][edge 2E][batch B*45][perm B*45][perm_lab B*90][mask E]
    float* out     = (float*)d_out;
    float* o_x     = out;
    float* o_edge  = o_x    + (long)B * KKEEP * D;
    float* o_batch = o_edge + 2 * E;
    float* o_perm  = o_batch + (long)B * KKEEP;
    float* o_plab  = o_perm  + (long)B * KKEEP;
    float* o_mask  = o_plab  + (long)B * MM;

    // workspace: [wnorm pad16][scores N f32][node_map N i32][perm B*45 i32]
    char*  ws       = (char*)d_ws;
    float* wnorm    = (float*)ws;
    float* scores   = (float*)(ws + 16);
    int*   node_map = (int*)(ws + 16 + N * sizeof(float));
    int*   perm_int = (int*)(ws + 16 + N * sizeof(float) + N * sizeof(int));

    k_wnorm<<<1, 128, 0, stream>>>(w, D, wnorm);

    long tiles  = (N + 15) / 16;
    long blocks = (tiles + 7) / 8;                 // 8 waves (tiles) per block
    k_scores<<<dim3((unsigned)blocks), dim3(256), 0, stream>>>(x, w, wnorm, scores, N, D);

    k_pool<<<dim3((unsigned)B), dim3(128), 0, stream>>>(scores, node_map, perm_int,
                                                        o_batch, o_perm, o_plab);

    long gth = (long)B * KKEEP * (D / 4);
    k_gather<<<dim3((unsigned)((gth + 255) / 256)), dim3(256), 0, stream>>>(
        x, scores, perm_int, o_x, D);

    k_edges<<<dim3((unsigned)((E + 255) / 256)), dim3(256), 0, stream>>>(
        ei, node_map, E, o_edge, o_mask);
}
